// EarthAttention3D_21552145891645
// MI455X (gfx1250) — matrix operand order repaired and checked by reading, not compiled, not run on hardware
//
#include <hip/hip_runtime.h>
#include <hip/hip_bf16.h>

typedef __attribute__((ext_vector_type(16))) _Float16 v16h;
typedef __attribute__((ext_vector_type(8)))  _Float16 v8h;
typedef __attribute__((ext_vector_type(8)))  float    v8f;

union A16 { v16h v; v8h h8[2]; };

#define DIM_C 192
#define NHEAD 6
#define HD 32
#define NWIN 144
#define BWIN 930
#define NTYPE 240
#define BIAS_SZ 3312
#define TOK (BWIN * NWIN)          // 133920
#define SCALE_Q 0.17677669529663687f

__device__ __forceinline__ v8f wmma16(v16h a, v16h b, v8f c) {
    return __builtin_amdgcn_wmma_f32_16x16x32_f16(false, a, false, b, (short)0, c, false, false);
}

// ---------------- fp32 -> f16 conversion ----------------
__global__ void cvt_f16_kernel(const float* __restrict__ src, _Float16* __restrict__ dst, int n) {
    int i = blockIdx.x * blockDim.x + threadIdx.x;
    int stride = gridDim.x * blockDim.x;
    for (; i < n; i += stride) dst[i] = (_Float16)src[i];
}

// ---------------- bias: mean over TYPE axis -> bias_red[3312][6] ----------------
__global__ void bias_reduce_kernel(const float* __restrict__ table, float* __restrict__ red) {
    int t = blockIdx.x * blockDim.x + threadIdx.x;
    if (t >= BIAS_SZ * NHEAD) return;
    int idx = t / NHEAD, h = t % NHEAD;
    const float* src = table + (size_t)idx * NTYPE * NHEAD + h;
    float s = 0.f;
    for (int k = 0; k < NTYPE; ++k) s += src[k * NHEAD];
    red[t] = s * (1.0f / (float)NTYPE);
}

// ---------------- bias: expand to bias_full[H][N][N] via position index ----------------
__global__ void bias_expand_kernel(const float* __restrict__ red, float* __restrict__ full) {
    int t = blockIdx.x * blockDim.x + threadIdx.x;
    if (t >= NHEAD * NWIN * NWIN) return;
    int h = t / (NWIN * NWIN);
    int r = t % (NWIN * NWIN);
    int n = r / NWIN, m = r % NWIN;
    int z1 = n / 72, y1 = (n % 72) / 12, x1 = n % 12;
    int z2 = m / 72, y2 = (m % 72) / 12, x2 = m % 12;
    int idx = (x1 - x2 + 11) + (y1 - y2) * 23 + (z1 - z2) * 828;
    if (idx < 0) idx += BIAS_SZ;   // JAX negative-index wrap
    full[t] = red[idx * NHEAD + h];
}

// ---------------- QKV GEMM: (TOK x 192) x (192 x 576), f16 WMMA ----------------
// grid: TOK/32 blocks of 256 threads (8 waves). wave -> (mblock half, 144-col group)
__global__ __launch_bounds__(256) void qkv_gemm_kernel(
        const _Float16* __restrict__ xh, const _Float16* __restrict__ wh,
        const float* __restrict__ qkv_b,
        _Float16* __restrict__ qb, _Float16* __restrict__ kb, _Float16* __restrict__ vb) {
    int wave = threadIdx.x >> 5, lane = threadIdx.x & 31;
    int mblock = blockIdx.x * 2 + (wave >> 2);
    int ngrp = wave & 3;                 // columns [ngrp*144, ngrp*144+144)
    int col = lane & 15, half = lane >> 4;
    int kb8 = half * 8, kb16 = half * 16, mbase = half * 8;
    int trow = mblock * 16 + col;

    v8f acc[9] = {};
#pragma unroll
    for (int ks = 0; ks < 6; ++ks) {
        const _Float16* ap = xh + (size_t)trow * DIM_C + ks * 32;
        A16 a;
        a.h8[0] = *(const v8h*)(ap + kb16);
        a.h8[1] = *(const v8h*)(ap + kb16 + 8);
#pragma unroll
        for (int t = 0; t < 9; ++t) {
            int j = ngrp * 144 + t * 16 + col;
            v16h bvec = *(const v16h*)(wh + (size_t)j * DIM_C + ks * 32 + kb16);
            acc[t] = wmma16(a.v, bvec, acc[t]);
        }
    }
    // epilogue: +bias, scatter into q (scaled) / k / v at [b][h][n][32] f16
#pragma unroll
    for (int t = 0; t < 9; ++t) {
        int j = ngrp * 144 + t * 16 + col;
        float bj = qkv_b[j];
        int which = j / DIM_C, c = j % DIM_C;
        int hh = c >> 5, d = c & 31;
#pragma unroll
        for (int i = 0; i < 8; ++i) {
            int tok = mblock * 16 + mbase + i;
            int bw = tok / NWIN, n = tok % NWIN;
            float val = acc[t][i] + bj;
            size_t off = (((size_t)(bw * NHEAD + hh)) * NWIN + n) * HD + d;
            if (which == 0)      qb[off] = (_Float16)(val * SCALE_Q);
            else if (which == 1) kb[off] = (_Float16)val;
            else                 vb[off] = (_Float16)val;
        }
    }
}

// ---------------- fused attention per (b,h): S, bias+mask, softmax, P*V ----------------
__global__ __launch_bounds__(288) void attn_kernel(
        const _Float16* __restrict__ qb, const _Float16* __restrict__ kb,
        const _Float16* __restrict__ vb, const float* __restrict__ mask,
        const float* __restrict__ bias_full, _Float16* __restrict__ obuf) {
    __shared__ _Float16 vs[160 * HD];          // V, rows 144..159 zero-padded
    __shared__ _Float16 Pl[9][16 * 160];       // per-wave P rows, cols 144..159 zero

    int bh = blockIdx.x;
    int bw = bh / NHEAD, h = bh % NHEAD;
    int tid = threadIdx.x, wave = tid >> 5, lane = tid & 31;
    int col = lane & 15, half = lane >> 4;
    int kb8 = half * 8, kb16 = half * 16, mbase = half * 8;

    // stage V into LDS (coalesced 16B copies) + zero pad rows
    {
        const v8h* vsrc = (const v8h*)(vb + (size_t)bh * NWIN * HD);
        v8h* vdst = (v8h*)vs;
        for (int p = tid; p < (NWIN * HD) / 8; p += 288) vdst[p] = vsrc[p];
        v8h zz = {};
        for (int p = tid; p < (16 * HD) / 8; p += 288) ((v8h*)(vs + NWIN * HD))[p] = zz;
    }
    __syncthreads();

    // S = q * k^T : 9 tiles, single K-step of 32 (hd == 32)
    A16 a;
    {
        const _Float16* qrow = qb + ((size_t)bh * NWIN + wave * 16 + col) * HD;
        a.h8[0] = *(const v8h*)(qrow + kb16);
        a.h8[1] = *(const v8h*)(qrow + kb16 + 8);
    }
    v8f s[9];
#pragma unroll
    for (int t = 0; t < 9; ++t) {
        v16h bvec = *(const v16h*)(kb + ((size_t)bh * NWIN + t * 16 + col) * HD + kb16);
        v8f z = {};
        s[t] = wmma16(a.v, bvec, z);
    }

    // + earth-position bias + window mask
    const float* mrow = mask + (size_t)bw * NWIN * NWIN;
    const float* brow = bias_full + (size_t)h * NWIN * NWIN;
#pragma unroll
    for (int t = 0; t < 9; ++t) {
        int nk = t * 16 + col;
#pragma unroll
        for (int i = 0; i < 8; ++i) {
            int nq = wave * 16 + mbase + i;
            s[t][i] += brow[nq * NWIN + nk] + mrow[nq * NWIN + nk];
        }
    }

    // row softmax (rows split across half-waves; reduce over 16 lanes) -> P in LDS f16
    _Float16* Pw = Pl[wave];
#pragma unroll
    for (int i = 0; i < 8; ++i) {
        float m = s[0][i];
#pragma unroll
        for (int t = 1; t < 9; ++t) m = fmaxf(m, s[t][i]);
#pragma unroll
        for (int off = 1; off < 16; off <<= 1) m = fmaxf(m, __shfl_xor(m, off, 16));
        float sum = 0.f;
#pragma unroll
        for (int t = 0; t < 9; ++t) { float e = __expf(s[t][i] - m); s[t][i] = e; sum += e; }
#pragma unroll
        for (int off = 1; off < 16; off <<= 1) sum += __shfl_xor(sum, off, 16);
        float inv = 1.0f / sum;
        int r = mbase + i;
#pragma unroll
        for (int t = 0; t < 9; ++t) Pw[r * 160 + t * 16 + col] = (_Float16)(s[t][i] * inv);
        Pw[r * 160 + 144 + col] = (_Float16)0.f;   // zero K-pad
    }

    // O = P * V : K = 160 (padded) -> 5 WMMA steps, 2 N-tiles of 16 dims
#pragma unroll
    for (int nt = 0; nt < 2; ++nt) {
        v8f o = {};
        int dim = nt * 16 + col;
#pragma unroll
        for (int ks = 0; ks < 5; ++ks) {
            A16 pa;
            const _Float16* pp = Pw + col * 160 + ks * 32;
            pa.h8[0] = *(const v8h*)(pp + kb8);
            pa.h8[1] = *(const v8h*)(pp + 16 + kb8);
            v16h bv;
#pragma unroll
            for (int i = 0; i < 16; ++i) bv[i] = vs[(ks * 32 + kb8 + i + (i & 8)) * HD + dim];
            o = wmma16(pa.v, bv, o);
        }
#pragma unroll
        for (int i = 0; i < 8; ++i) {
            int n = wave * 16 + mbase + i;
            obuf[((size_t)(bw * NWIN + n)) * DIM_C + h * HD + nt * 16 + col] = (_Float16)o[i];
        }
    }
}

// ---------------- proj GEMM: (TOK x 192) x (192 x 192) + bias -> fp32 out ----------------
__global__ __launch_bounds__(256) void proj_gemm_kernel(
        const _Float16* __restrict__ obuf, const _Float16* __restrict__ wh,
        const float* __restrict__ proj_b, float* __restrict__ out) {
    int wave = threadIdx.x >> 5, lane = threadIdx.x & 31;
    int mblock = blockIdx.x * 2 + (wave >> 2);
    int ngrp = wave & 3;                // 48 columns each
    int col = lane & 15, half = lane >> 4;
    int kb8 = half * 8, kb16 = half * 16, mbase = half * 8;
    int trow = mblock * 16 + col;

    v8f acc[3] = {};
#pragma unroll
    for (int ks = 0; ks < 6; ++ks) {
        const _Float16* ap = obuf + (size_t)trow * DIM_C + ks * 32;
        A16 a;
        a.h8[0] = *(const v8h*)(ap + kb16);
        a.h8[1] = *(const v8h*)(ap + kb16 + 8);
#pragma unroll
        for (int t = 0; t < 3; ++t) {
            int j = ngrp * 48 + t * 16 + col;
            v16h bvec = *(const v16h*)(wh + (size_t)j * DIM_C + ks * 32 + kb16);
            acc[t] = wmma16(a.v, bvec, acc[t]);
        }
    }
#pragma unroll
    for (int t = 0; t < 3; ++t) {
        int j = ngrp * 48 + t * 16 + col;
        float bj = proj_b[j];
#pragma unroll
        for (int i = 0; i < 8; ++i) {
            int tok = mblock * 16 + mbase + i;
            out[(size_t)tok * DIM_C + j] = acc[t][i] + bj;
        }
    }
}

extern "C" void kernel_launch(void* const* d_in, const int* in_sizes, int n_in,
                              void* d_out, int out_size, void* d_ws, size_t ws_size,
                              hipStream_t stream) {
    const float* x          = (const float*)d_in[0];
    const float* mask       = (const float*)d_in[1];
    const float* qkv_w      = (const float*)d_in[2];
    const float* qkv_b      = (const float*)d_in[3];
    const float* proj_w     = (const float*)d_in[4];
    const float* proj_b     = (const float*)d_in[5];
    const float* bias_table = (const float*)d_in[6];
    float* out = (float*)d_out;

    // workspace partition (256B aligned)
    char* p = (char*)d_ws;
    auto alloc = [&](size_t bytes) -> char* {
        char* r = p; p += (bytes + 255) & ~(size_t)255; return r;
    };
    _Float16* xh      = (_Float16*)alloc((size_t)TOK * DIM_C * 2);   // reused as obuf later
    _Float16* qkv_wh  = (_Float16*)alloc((size_t)576 * DIM_C * 2);
    _Float16* proj_wh = (_Float16*)alloc((size_t)DIM_C * DIM_C * 2);
    _Float16* qbuf    = (_Float16*)alloc((size_t)BWIN * NHEAD * NWIN * HD * 2);
    _Float16* kbuf    = (_Float16*)alloc((size_t)BWIN * NHEAD * NWIN * HD * 2);
    _Float16* vbuf    = (_Float16*)alloc((size_t)BWIN * NHEAD * NWIN * HD * 2);
    float*    bias_red  = (float*)alloc((size_t)BIAS_SZ * NHEAD * 4);
    float*    bias_full = (float*)alloc((size_t)NHEAD * NWIN * NWIN * 4);
    _Float16* obuf = xh;   // x no longer needed once attention runs

    // 1) conversions
    cvt_f16_kernel<<<2048, 256, 0, stream>>>(x, xh, TOK * DIM_C);
    cvt_f16_kernel<<<108, 256, 0, stream>>>(qkv_w, qkv_wh, 576 * DIM_C);
    cvt_f16_kernel<<<36, 256, 0, stream>>>(proj_w, proj_wh, DIM_C * DIM_C);

    // 2) bias precompute
    bias_reduce_kernel<<<(BIAS_SZ * NHEAD + 255) / 256, 256, 0, stream>>>(bias_table, bias_red);
    bias_expand_kernel<<<(NHEAD * NWIN * NWIN + 255) / 256, 256, 0, stream>>>(bias_red, bias_full);

    // 3) QKV projection (TOK/32 = 4185 workgroups)
    qkv_gemm_kernel<<<TOK / 32, 256, 0, stream>>>(xh, qkv_wh, qkv_b, qbuf, kbuf, vbuf);

    // 4) fused attention: one workgroup per (window, head)
    attn_kernel<<<BWIN * NHEAD, 288, 0, stream>>>(qbuf, kbuf, vbuf, mask, bias_full, obuf);

    // 5) output projection
    proj_gemm_kernel<<<TOK / 32, 256, 0, stream>>>(obuf, proj_wh, proj_b, out);
}
